// XcodecEuclideanCodebook_7636451852798
// MI455X (gfx1250) — matrix operand order repaired and checked by reading, not compiled, not run on hardware
//
#include <hip/hip_runtime.h>

// ---------------------------------------------------------------------------
// VQ codebook nearest-neighbor (XcodecEuclideanCodebook) for gfx1250 / MI455X
//   score[n] = 2 * <x, e_n> - ||e_n||^2   (x_sq is row-constant -> dropped)
//   Prep kernel: codebook f32 -> bf16 (+ ||e||^2) once into d_ws.
//   Main kernel: d_ws -> LDS via global_load_async_to_lds_b128 (ASYNCcnt),
//   then per-wave v_wmma_f32_16x16x32_bf16 GEMM with double-buffered B tiles,
//   fused argmax, and coalesced f32 codeword gather.
// ---------------------------------------------------------------------------

typedef __attribute__((ext_vector_type(16))) __bf16     v16bf;
typedef __attribute__((ext_vector_type(8)))  float      v8f;
typedef __attribute__((ext_vector_type(4)))  float      f4;
typedef __attribute__((ext_vector_type(4)))  unsigned   u32x4;

union V16U { v16bf bf; u32x4 u[2]; };

constexpr int K_CB      = 1024;          // codebook size
constexpr int D         = 128;           // codeword dim
constexpr int ROWSTRIDE = 136;           // bf16 row stride (+8 pad -> conflict-free ds_load_b128)
constexpr int EB_ELEMS  = K_CB * ROWSTRIDE;
constexpr size_t EB_BYTES   = (size_t)EB_ELEMS * 2;            // 278528
constexpr size_t SMEM_BYTES = EB_BYTES + (size_t)K_CB * 4;     // 282624 (16B multiple)
constexpr int BLOCK     = 512;           // 16 waves (wave32)
constexpr int WAVES     = BLOCK / 32;

// ---------------- prep: build bf16 codebook + ||e||^2 in d_ws ---------------
__global__ __launch_bounds__(256)
void vq_prep_kernel(const float* __restrict__ embed, char* __restrict__ ws)
{
    __bf16* eb  = (__bf16*)ws;
    float*  esq = (float*)(ws + EB_BYTES);
    const int tid  = blockIdx.x * blockDim.x + threadIdx.x;
    const int nthr = gridDim.x * blockDim.x;

    for (int c = tid; c < (K_CB * D) / 4; c += nthr) {
        const int base = c * 4;
        const int row  = base >> 7;
        const int col  = base & (D - 1);
        f4 v = *(const f4*)(embed + base);
        __bf16* dst = eb + row * ROWSTRIDE + col;
        dst[0] = (__bf16)v.x; dst[1] = (__bf16)v.y;
        dst[2] = (__bf16)v.z; dst[3] = (__bf16)v.w;
    }
    for (int r = tid; r < K_CB; r += nthr) {
        const float* er = embed + (size_t)r * D;
        float s = 0.f;
        for (int c = 0; c < D; ++c) {
            const float v = (float)(__bf16)er[c];   // same quantization as GEMM
            s += v * v;
        }
        esq[r] = s;
    }
}

// ------------------------------- main kernel --------------------------------
__device__ __forceinline__ void load_btile(V16U* B, const __bf16* eb,
                                           int nb, int lanelo, int g) {
    // B operand, contiguous K order used here (32x16): lane holds col (lane&15); 16 consecutive K
    // starting at ks*32 + g*16  ->  2x ds_load_b128 per K-step.
    const __bf16* brow = eb + (size_t)(nb + lanelo) * ROWSTRIDE + g * 16;
    #pragma unroll
    for (int ks = 0; ks < 4; ++ks) {
        const __bf16* bp = brow + ks * 32;
        B[ks].u[0] = *(const u32x4*)bp;
        B[ks].u[1] = *(const u32x4*)(bp + 8);
    }
}

__device__ __forceinline__ v8f wmma_chain(const v16bf* A, const V16U* B) {
    v8f acc = {};
    #pragma unroll
    for (int ks = 0; ks < 4; ++ks)
        acc = __builtin_amdgcn_wmma_f32_16x16x32_bf16(
                false, A[ks], false, B[ks].bf, (short)0, acc, false, false);
    return acc;
}

__device__ __forceinline__ void argmax_update(const v8f& acc, float en, int n,
                                              float* best, int* bidx) {
    #pragma unroll
    for (int r = 0; r < 8; ++r) {
        const float sc = __builtin_fmaf(acc[r], 2.f, -en);  // lane: rows r / r+8
        const bool take = sc > best[r];
        best[r] = take ? sc : best[r];
        bidx[r] = take ? n  : bidx[r];
    }
}

__global__ __launch_bounds__(BLOCK)
void vq_wmma_kernel(const float* __restrict__ x,
                    const float* __restrict__ embed,
                    const char*  __restrict__ ws,
                    int*   __restrict__ ind_out,
                    float* __restrict__ q_out,
                    int n_rows)
{
    extern __shared__ char smem[];                       // dynamic LDS base == offset 0
    const __bf16* eb  = (const __bf16*)smem;             // [K_CB][ROWSTRIDE] bf16
    const float*  esq = (const float*)(smem + EB_BYTES); // [K_CB] ||e_bf16||^2

    const int t = threadIdx.x;

    // ---- Phase 1: bulk-copy prebuilt codebook image ws -> LDS, async ----
    {
        constexpr int CHUNKS = (int)(SMEM_BYTES / 16);   // 17664 x 16B
        const unsigned long long gbase = (unsigned long long)ws;
        for (int c = t; c < CHUNKS; c += BLOCK) {
            const unsigned           lds_off = (unsigned)c * 16u;
            const unsigned long long gaddr   = gbase + (unsigned long long)c * 16u;
            asm volatile("global_load_async_to_lds_b128 %0, %1, off"
                         :: "v"(lds_off), "v"(gaddr) : "memory");
        }
        asm volatile("s_wait_asynccnt 0" ::: "memory");
    }
    __syncthreads();

    // ---- Phase 2: per-wave 16-row tiles, pipelined WMMA GEMM + fused argmax ----
    const int lane   = t & 31;
    const int lanelo = lane & 15;
    const int g      = lane >> 4;          // lane half (K-group select in WMMA layout)
    const int wave   = t >> 5;
    const int ntiles = n_rows >> 4;

    for (int tile = blockIdx.x * WAVES + wave; tile < ntiles;
         tile += gridDim.x * WAVES) {
        const int mbase = tile * 16;

        // Prefetch next tile's x rows (global_prefetch_b8).
        {
            const int nt = tile + gridDim.x * WAVES;
            if (nt < ntiles)
                __builtin_prefetch(x + (size_t)(nt * 16 + lanelo) * D, 0, 1);
        }

        // Load A (16x128 f32 -> bf16), A, contiguous K order used here: lane holds row
        // (lane&15); elems 0..15 = K = g*16+0..15 (matches B).
        v16bf A[4];
        const float* xr = x + (size_t)(mbase + lanelo) * D;
        #pragma unroll
        for (int ks = 0; ks < 4; ++ks) {
            const int cb = ks * 32 + g * 16;
            f4 p0 = *(const f4*)(xr + cb);
            f4 p1 = *(const f4*)(xr + cb + 4);
            f4 q0 = *(const f4*)(xr + cb + 8);
            f4 q1 = *(const f4*)(xr + cb + 12);
            A[ks][0]  = (__bf16)p0.x; A[ks][1]  = (__bf16)p0.y;
            A[ks][2]  = (__bf16)p0.z; A[ks][3]  = (__bf16)p0.w;
            A[ks][4]  = (__bf16)p1.x; A[ks][5]  = (__bf16)p1.y;
            A[ks][6]  = (__bf16)p1.z; A[ks][7]  = (__bf16)p1.w;
            A[ks][8]  = (__bf16)q0.x; A[ks][9]  = (__bf16)q0.y;
            A[ks][10] = (__bf16)q0.z; A[ks][11] = (__bf16)q0.w;
            A[ks][12] = (__bf16)q1.x; A[ks][13] = (__bf16)q1.y;
            A[ks][14] = (__bf16)q1.z; A[ks][15] = (__bf16)q1.w;
        }

        float best[8];
        int   bidx[8];
        #pragma unroll
        for (int r = 0; r < 8; ++r) { best[r] = -3.4e38f; bidx[r] = 0; }

        // Double-buffered B, two independent accumulation chains per iteration.
        // Prefetch indices are clamped (uniform s_cselect) -> branch-free body.
        V16U B0[4], B1[4];
        load_btile(B0, eb, 0,  lanelo, g);
        load_btile(B1, eb, 16, lanelo, g);

        for (int nb = 0; nb < K_CB; nb += 32) {
            const int nbn0 = (nb + 32 < K_CB) ? nb + 32 : K_CB - 16;
            const int nbn1 = (nb + 48 < K_CB) ? nb + 48 : K_CB - 16;

            const v8f acc0 = wmma_chain(A, B0);            // tile nb
            load_btile(B0, eb, nbn0, lanelo, g);
            const v8f acc1 = wmma_chain(A, B1);            // tile nb+16
            load_btile(B1, eb, nbn1, lanelo, g);

            const float en0 = esq[nb + lanelo];
            const float en1 = esq[nb + 16 + lanelo];
            argmax_update(acc0, en0, nb + lanelo,      best, bidx);
            argmax_update(acc1, en1, nb + 16 + lanelo, best, bidx);
        }

        // Branchless cross-lane argmax within each 16-lane half (rows 0..7 / 8..15).
        #pragma unroll
        for (int off = 8; off >= 1; off >>= 1) {
            #pragma unroll
            for (int r = 0; r < 8; ++r) {
                const float ov = __shfl_xor(best[r], off, 32);
                const int   oi = __shfl_xor(bidx[r], off, 32);
                const bool take = (ov > best[r]) | ((ov == best[r]) & (oi < bidx[r]));
                best[r] = take ? ov : best[r];
                bidx[r] = take ? oi : bidx[r];
            }
        }

        // Lanes 0 / 16 hold winners for rows mbase+r / mbase+8+r.
        if (lanelo == 0) {
            const int half = g * 8;
            #pragma unroll
            for (int r = 0; r < 8; ++r)
                ind_out[mbase + half + r] = bidx[r];
        }

        // Gather quantized rows from the ORIGINAL f32 codebook (matches reference).
        #pragma unroll
        for (int m = 0; m < 16; ++m) {
            const int src = __shfl(bidx[m & 7], (m < 8) ? 0 : 16, 32);
            f4 v = *(const f4*)(embed + (size_t)src * D + lane * 4);
            *(f4*)(q_out + (size_t)(mbase + m) * D + lane * 4) = v;
        }
    }
}

extern "C" void kernel_launch(void* const* d_in, const int* in_sizes, int n_in,
                              void* d_out, int out_size, void* d_ws, size_t ws_size,
                              hipStream_t stream) {
    const float* x     = (const float*)d_in[0];   // [8,4096,128] f32
    const float* embed = (const float*)d_in[1];   // [1024,128]  f32
    const int n_rows = in_sizes[0] / D;           // 32768

    int*   ind = (int*)d_out;                     // output 0: indices (int32)
    float* q   = (float*)d_out + n_rows;          // output 1: quantized f32

    // 1) Build bf16 codebook image (+ ||e||^2) once in workspace.
    vq_prep_kernel<<<64, 256, 0, stream>>>(embed, (char*)d_ws);

    // 2) Main fused kernel.
    static_cast<void>(hipFuncSetAttribute(
        reinterpret_cast<const void*>(vq_wmma_kernel),
        hipFuncAttributeMaxDynamicSharedMemorySize, (int)SMEM_BYTES));

    int ntiles = n_rows >> 4;                     // 2048
    int blocks = (ntiles + WAVES - 1) / WAVES;    // 128 -> 1 tile per wave
    if (blocks > 128) blocks = 128;
    if (blocks < 1)   blocks = 1;

    vq_wmma_kernel<<<blocks, BLOCK, SMEM_BYTES, stream>>>(
        x, embed, (const char*)d_ws, ind, q, n_rows);
}
